// BidirectionalRead_15126874816737
// MI455X (gfx1250) — matrix operand order repaired and checked by reading, not compiled, not run on hardware
//
#include <hip/hip_runtime.h>

// ---------------------------------------------------------------------------
// Problem constants (N=2, C=256, H=W=64, HW=4096, Cl=16)
// ---------------------------------------------------------------------------
#define HW   4096
#define CCH  256
#define CHW  (CCH * HW)          // 1048576
#define NB   2
#define CL   16

// Workspace byte offsets (total ~178 MB, all 256B aligned)
#define WS_LOGITS1 ((size_t)0)              //  4*4096*4      = 64 KB
#define WS_LOGITS2 ((size_t)65536)          //  64 KB
#define WS_FCL     ((size_t)131072)         //  2*2*16*4096*4 = 1 MB
#define WS_FSA     ((size_t)1179648)        //  2*2*256*4096*2 = 8 MB (bf16)
#define WS_ABF     ((size_t)9568256)        //  2*4096*4096*2 = 64 MB (bf16)
#define WS_ATBF    ((size_t)76677120)       //  64 MB (bf16)
#define WS_FHAT    ((size_t)143785984)      //  2*2*256*4096*4 = 16 MB
#define WS_FP      ((size_t)160563200)      //  16 MB  -> end 177340416

typedef __attribute__((ext_vector_type(16))) __bf16 bf16x16;
typedef __attribute__((ext_vector_type(8)))  float  fx8;

union OpU { bf16x16 v; unsigned short h[16]; uint4 q[2]; };

static __device__ __forceinline__ unsigned short f2bf(float f) {
  union { float f; unsigned u; } x; x.f = f;
  unsigned u = x.u;
  unsigned r = u + 0x7fffu + ((u >> 16) & 1u);   // round-to-nearest-even
  return (unsigned short)(r >> 16);
}

#if defined(__has_builtin)
#if __has_builtin(__builtin_amdgcn_tanhf)
#define TANHF(x) __builtin_amdgcn_tanhf(x)
#else
#define TANHF(x) tanhf(x)
#endif
#else
#define TANHF(x) tanhf(x)
#endif

static __device__ __forceinline__ fx8 wmma_bf16(bf16x16 a, bf16x16 b, fx8 c) {
  return __builtin_amdgcn_wmma_f32_16x16x32_bf16(
      /*neg_a=*/false, a, /*neg_b=*/false, b,
      /*c_mod=*/(short)0, c, /*reuse_a=*/false, /*reuse_b=*/false);
}

// ---------------------------------------------------------------------------
// K1/K7-helper: logits[s][n][p] = sum_c f[n,c,p]*cw_s[c] + cb_s
// ---------------------------------------------------------------------------
__global__ void k_logits(const float* __restrict__ fA, const float* __restrict__ fB,
                         const float* __restrict__ cw1, const float* __restrict__ cb1,
                         const float* __restrict__ cw2, const float* __restrict__ cb2,
                         float* __restrict__ logits) {
  int t = blockIdx.x * blockDim.x + threadIdx.x;      // 16384 total
  if (t >= 2 * NB * HW) return;
  int p = t & (HW - 1);
  int n = (t >> 12) & 1;
  int s = t >> 13;
  const float* f  = (s ? fB : fA) + (size_t)n * CHW + p;
  const float* cw = s ? cw2 : cw1;
  float acc = 0.f;
  for (int c = 0; c < CCH; ++c) acc += f[(size_t)c * HW] * cw[c];
  logits[t] = acc + (s ? cb2[0] : cb1[0]);
}

// ---------------------------------------------------------------------------
// Softmax over 4096 elements; one block (256 thr) per row. In-place -> probs.
// ---------------------------------------------------------------------------
__global__ void k_softmax(float* __restrict__ buf) {
  __shared__ float red[256];
  float* row = buf + (size_t)blockIdx.x * HW;
  int t = threadIdx.x;
  float m = -3.4e38f;
  for (int i = t; i < HW; i += 256) m = fmaxf(m, row[i]);
  red[t] = m; __syncthreads();
  for (int o = 128; o > 0; o >>= 1) { if (t < o) red[t] = fmaxf(red[t], red[t + o]); __syncthreads(); }
  m = red[0]; __syncthreads();
  float sum = 0.f;
  for (int i = t; i < HW; i += 256) { float e = expf(row[i] - m); row[i] = e; sum += e; }
  red[t] = sum; __syncthreads();
  for (int o = 128; o > 0; o >>= 1) { if (t < o) red[t] += red[t + o]; __syncthreads(); }
  float inv = 1.f / red[0];
  for (int i = t; i < HW; i += 256) row[i] *= inv;
}

// ---------------------------------------------------------------------------
// K3a: fsa_bf16[s][n][c][p] = bf16( sc[s][n][p] * f[n][c][p] )   (4 elem/thr)
// ---------------------------------------------------------------------------
__global__ void k_scale_bf16(const float* __restrict__ f1, const float* __restrict__ f2,
                             const float* __restrict__ sc, unsigned short* __restrict__ fsa) {
  int t = blockIdx.x * blockDim.x + threadIdx.x;     // 1,048,576 threads
  size_t idx4 = (size_t)t * 4;
  int p = idx4 & (HW - 1);
  int c = (idx4 >> 12) & 255;
  int n = (idx4 >> 20) & 1;
  int s = (int)(idx4 >> 21);
  const float* f   = (s ? f2 : f1) + (size_t)n * CHW + (size_t)c * HW + p;
  const float* scp = sc + (size_t)(s * NB + n) * HW + p;
  float4 v = *(const float4*)f;
  float4 w = *(const float4*)scp;
  union { unsigned short h[4]; uint2 u; } o;
  o.h[0] = f2bf(v.x * w.x); o.h[1] = f2bf(v.y * w.y);
  o.h[2] = f2bf(v.z * w.z); o.h[3] = f2bf(v.w * w.w);
  *(uint2*)(fsa + idx4) = o.u;
}

// ---------------------------------------------------------------------------
// K3b: f_cl[s][n][cl][p] = sc[p] * (sum_c pw_s[cl,c]*f[n,c,p]) + pb_s[cl]
// ---------------------------------------------------------------------------
__global__ void k_project(const float* __restrict__ f1, const float* __restrict__ f2,
                          const float* __restrict__ sc,
                          const float* __restrict__ pw1, const float* __restrict__ pb1,
                          const float* __restrict__ pw2, const float* __restrict__ pb2,
                          float* __restrict__ fcl) {
  int t = blockIdx.x * blockDim.x + threadIdx.x;     // 16384
  if (t >= 2 * NB * HW) return;
  int p = t & (HW - 1);
  int n = (t >> 12) & 1;
  int s = t >> 13;
  const float* f  = (s ? f2 : f1) + (size_t)n * CHW + p;
  const float* pw = s ? pw2 : pw1;
  const float* pb = s ? pb2 : pb1;
  float acc[CL];
#pragma unroll
  for (int l = 0; l < CL; ++l) acc[l] = 0.f;
  for (int c = 0; c < CCH; ++c) {
    float v = f[(size_t)c * HW];
#pragma unroll
    for (int l = 0; l < CL; ++l) acc[l] += pw[l * CCH + c] * v;
  }
  float scale = sc[(size_t)(s * NB + n) * HW + p];
  float* o = fcl + ((size_t)(s * NB + n) * CL) * HW + p;
#pragma unroll
  for (int l = 0; l < CL; ++l) o[(size_t)l * HW] = scale * acc[l] + pb[l];
}

// ---------------------------------------------------------------------------
// K4: A[n][p][q] = tanh( sum_cl f2_cl[cl,p] * f1_cl[cl,q] ) via WMMA (K=16
// zero-padded to 32). One wave per 16x16 tile. Writes A (row-major [p][q])
// and A^T (row-major [q][p]) in bf16 so both big GEMMs get contiguous B.
// ---------------------------------------------------------------------------
__global__ void k_corr_wmma(const float* __restrict__ fcl,
                            unsigned short* __restrict__ Abf,
                            unsigned short* __restrict__ ATbf) {
  int w    = blockIdx.x * (blockDim.x >> 5) + (threadIdx.x >> 5);  // 131072 waves
  int lane = threadIdx.x & 31;
  int nb  = w >> 16;
  int rem = w & 65535;
  int p0 = (rem >> 8) << 4;
  int q0 = (rem & 255) << 4;
  const float* fcl1 = fcl + (size_t)(0 * NB + nb) * CL * HW;  // f1_cl (q side)
  const float* fcl2 = fcl + (size_t)(1 * NB + nb) * CL * HW;  // f2_cl (p side)
  int m = lane & 15, h = lane >> 4;
  OpU a, b;
#pragma unroll
  for (int i = 0; i < 8; ++i) {                 // A-op: K=8h+i (<16 real), K>=16 zero
    a.h[i]     = f2bf(fcl2[(size_t)(8 * h + i) * HW + p0 + m]);
    a.h[8 + i] = 0;
  }
#pragma unroll
  for (int i = 0; i < 16; ++i)                  // B-op: K=8h+i for i<8 (<16 real), K>=16 zero
    b.h[i] = (i < 8) ? f2bf(fcl1[(size_t)(8 * h + i) * HW + q0 + m]) : (unsigned short)0;
  fx8 c = {};
  c = wmma_bf16(a.v, b.v, c);
  unsigned short* An  = Abf  + (size_t)nb * HW * HW;
  unsigned short* ATn = ATbf + (size_t)nb * HW * HW;
  int col = lane & 15;
#pragma unroll
  for (int r = 0; r < 8; ++r) {
    int row = 8 * h + r;
    unsigned short bv = f2bf(TANHF(c[r]));
    An [(size_t)(p0 + row) * HW + (q0 + col)] = bv;
    ATn[(size_t)(q0 + col) * HW + (p0 + row)] = bv;
  }
}

// ---------------------------------------------------------------------------
// K5: main GEMM. f_hat[s][n][c][q] = sum_p fsa[s][n][c][p] * Bmat[q][p]
//   s=0 (f1_hat): Bmat = A   ([q][p] row-major == A^T as B-operand)
//   s=1 (f2_hat): Bmat = A^T
// One wave computes a 64(row) x 16(col) strip: B tile loaded once per K-step,
// reused across 4 row sub-tiles (4 WMMAs / K-step, f32 accumulation).
// ---------------------------------------------------------------------------
__global__ void k_gemm_wmma(const unsigned short* __restrict__ fsa,
                            const unsigned short* __restrict__ Abf,
                            const unsigned short* __restrict__ ATbf,
                            float* __restrict__ fhat) {
  int w    = blockIdx.x * 4 + (threadIdx.x >> 5);   // 4096 waves
  int lane = threadIdx.x & 31;
  int s  = w >> 11;
  int n  = (w >> 10) & 1;
  int cg = (w >> 8) & 3;
  int q0 = (w & 255) << 4;
  int c0 = cg << 6;
  const unsigned short* fs = fsa + (size_t)(s * NB + n) * CHW;
  const unsigned short* Am = (s == 0 ? Abf : ATbf) + (size_t)n * HW * HW;
  int m = lane & 15, h = lane >> 4, col = lane & 15;
  fx8 acc0 = {}, acc1 = {}, acc2 = {}, acc3 = {};
  const unsigned short* brow  = Am + (size_t)(q0 + col) * HW + 8 * h;
  const unsigned short* arow0 = fs + (size_t)(c0 + m) * HW + 8 * h;
  for (int p = 0; p < HW; p += 32) {
    OpU b;
    b.q[0] = *(const uint4*)(brow + p);
    b.q[1] = *(const uint4*)(brow + p + 16);
    OpU a;
    a.q[0] = *(const uint4*)(arow0 + p);
    a.q[1] = *(const uint4*)(arow0 + p + 16);
    acc0 = wmma_bf16(a.v, b.v, acc0);
    a.q[0] = *(const uint4*)(arow0 + 16 * HW + p);
    a.q[1] = *(const uint4*)(arow0 + 16 * HW + p + 16);
    acc1 = wmma_bf16(a.v, b.v, acc1);
    a.q[0] = *(const uint4*)(arow0 + 32 * HW + p);
    a.q[1] = *(const uint4*)(arow0 + 32 * HW + p + 16);
    acc2 = wmma_bf16(a.v, b.v, acc2);
    a.q[0] = *(const uint4*)(arow0 + 48 * HW + p);
    a.q[1] = *(const uint4*)(arow0 + 48 * HW + p + 16);
    acc3 = wmma_bf16(a.v, b.v, acc3);
  }
  float* out = fhat + (size_t)(s * NB + n) * CHW;
#pragma unroll
  for (int r = 0; r < 8; ++r) {
    int row = 8 * h + r;
    out[(size_t)(c0 +  0 + row) * HW + q0 + col] = acc0[r];
    out[(size_t)(c0 + 16 + row) * HW + q0 + col] = acc1[r];
    out[(size_t)(c0 + 32 + row) * HW + q0 + col] = acc2[r];
    out[(size_t)(c0 + 48 + row) * HW + q0 + col] = acc3[r];
  }
}

// ---------------------------------------------------------------------------
// K6: per pixel: l2-normalize f_hat over channels, add residual, relu,
//     store fp, and accumulate second-pass logits.
// ---------------------------------------------------------------------------
__global__ void k_norm_res(const float* __restrict__ fhat,
                           const float* __restrict__ f1, const float* __restrict__ f2,
                           const float* __restrict__ cw1, const float* __restrict__ cb1,
                           const float* __restrict__ cw2, const float* __restrict__ cb2,
                           float* __restrict__ fp, float* __restrict__ logits2) {
  int t = blockIdx.x * blockDim.x + threadIdx.x;   // 16384
  if (t >= 2 * NB * HW) return;
  int p = t & (HW - 1);
  int n = (t >> 12) & 1;
  int s = t >> 13;
  const float* fh = fhat + (size_t)(s * NB + n) * CHW + p;
  const float* fo = (s ? f2 : f1) + (size_t)n * CHW + p;
  float ss = 0.f;
  for (int c = 0; c < CCH; ++c) { float v = fh[(size_t)c * HW]; ss += v * v; }
  float inv = 1.f / fmaxf(sqrtf(ss), 1e-12f);
  const float* cw = s ? cw2 : cw1;
  float* fpp = fp + (size_t)(s * NB + n) * CHW + p;
  float lg = 0.f;
  for (int c = 0; c < CCH; ++c) {
    float v = fmaxf(fh[(size_t)c * HW] * inv + fo[(size_t)c * HW], 0.f);
    fpp[(size_t)c * HW] = v;
    lg += cw[c] * v;
  }
  logits2[t] = lg + (s ? cb2[0] : cb1[0]);
}

// ---------------------------------------------------------------------------
// K8: out[s][n][c][p] = fp * softmax2 scale   (4 elem/thr)
// ---------------------------------------------------------------------------
__global__ void k_out(const float* __restrict__ fp, const float* __restrict__ sc2,
                      float* __restrict__ out) {
  int t = blockIdx.x * blockDim.x + threadIdx.x;   // 1,048,576 threads
  size_t idx4 = (size_t)t * 4;
  int p = idx4 & (HW - 1);
  int n = (idx4 >> 20) & 1;
  int s = (int)(idx4 >> 21);
  float4 v = *(const float4*)(fp + idx4);
  float4 w = *(const float4*)(sc2 + (size_t)(s * NB + n) * HW + p);
  float4 o; o.x = v.x * w.x; o.y = v.y * w.y; o.z = v.z * w.z; o.w = v.w * w.w;
  *(float4*)(out + idx4) = o;
}

// ---------------------------------------------------------------------------
extern "C" void kernel_launch(void* const* d_in, const int* in_sizes, int n_in,
                              void* d_out, int out_size, void* d_ws, size_t ws_size,
                              hipStream_t stream) {
  const float* f1  = (const float*)d_in[0];
  const float* f2  = (const float*)d_in[1];
  const float* pw1 = (const float*)d_in[2];
  const float* pb1 = (const float*)d_in[3];
  const float* pw2 = (const float*)d_in[4];
  const float* pb2 = (const float*)d_in[5];
  const float* cw1 = (const float*)d_in[6];
  const float* cb1 = (const float*)d_in[7];
  const float* cw2 = (const float*)d_in[8];
  const float* cb2 = (const float*)d_in[9];

  char* ws = (char*)d_ws;
  float*          lg1  = (float*)(ws + WS_LOGITS1);
  float*          lg2  = (float*)(ws + WS_LOGITS2);
  float*          fcl  = (float*)(ws + WS_FCL);
  unsigned short* fsa  = (unsigned short*)(ws + WS_FSA);
  unsigned short* Abf  = (unsigned short*)(ws + WS_ABF);
  unsigned short* ATbf = (unsigned short*)(ws + WS_ATBF);
  float*          fhat = (float*)(ws + WS_FHAT);
  float*          fp   = (float*)(ws + WS_FP);
  float*          out  = (float*)d_out;

  // Pass 1: SA logits + softmax
  k_logits   <<<64,    256, 0, stream>>>(f1, f2, cw1, cb1, cw2, cb2, lg1);
  k_softmax  <<<4,     256, 0, stream>>>(lg1);
  // bf16 scaled features + Cl=16 projections
  k_scale_bf16<<<4096, 256, 0, stream>>>(f1, f2, lg1, fsa);
  k_project  <<<64,    256, 0, stream>>>(f1, f2, lg1, pw1, pb1, pw2, pb2, fcl);
  // Correlation matrix A (and A^T) via WMMA + tanh
  k_corr_wmma<<<32768, 128, 0, stream>>>(fcl, Abf, ATbf);
  // Main GEMMs via WMMA
  k_gemm_wmma<<<1024,  128, 0, stream>>>(fsa, Abf, ATbf, fhat);
  // l2norm + residual + relu + pass-2 logits
  k_norm_res <<<64,    256, 0, stream>>>(fhat, f1, f2, cw1, cb1, cw2, cb2, fp, lg2);
  k_softmax  <<<4,     256, 0, stream>>>(lg2);
  // Final scaling into d_out ([f1_2 ; f2_2] flat)
  k_out      <<<4096,  256, 0, stream>>>(fp, lg2, out);
}
